// CrossViewTransformer_11536282157410
// MI455X (gfx1250) — matrix operand order repaired and checked by reading, not compiled, not run on hardware
//
#include <hip/hip_runtime.h>
#include <cstdint>

typedef __attribute__((ext_vector_type(16))) _Float16 v16h;
typedef __attribute__((ext_vector_type(8)))  float    v8f;
typedef __attribute__((ext_vector_type(4)))  unsigned int u32x4;
typedef __attribute__((ext_vector_type(8)))  int      i32x8;
typedef __attribute__((ext_vector_type(4)))  int      i32x4;

#define C_IN   128
#define C8     16
#define HW     4096
#define NTILE  256          // HW/16
#define KCONV  2304         // 9 taps * 256 ci ; K = (dy*3+dx)*256 + ci
#define KCH    72           // KCONV/32
#define CPAD   264          // padded ci stride in patch (halfs)

__device__ __forceinline__ uint32_t pack2(float a, float b) {
    union { _Float16 h[2]; uint32_t u; } x;
    x.h[0] = (_Float16)a; x.h[1] = (_Float16)b;
    return x.u;
}

// ---------------------------------------------------------------------------
// Kernel 1: Q = Wq*x_forward, K = Wk*x ; write f16 WMMA fragments.
//   K -> A-fragment (16 kpos x 32 K, K=16..31 zero pad)
//   Q -> B-fragment (32 K x 16 qpos, K=16..31 zero pad)
// storage: [b][tile][lane][8 dwords]
// ---------------------------------------------------------------------------
__global__ void proj_qk_kernel(const float* __restrict__ x,
                               const float* __restrict__ xf,
                               const float* __restrict__ Wq,
                               const float* __restrict__ bq,
                               const float* __restrict__ Wk,
                               const float* __restrict__ bk,
                               uint32_t* __restrict__ kfrag,
                               uint32_t* __restrict__ qfrag) {
    int pg  = blockIdx.x * blockDim.x + threadIdx.x;   // 16384 threads
    int b   = pg >> 12;
    int pos = pg & (HW - 1);

    float q[C8], k[C8];
#pragma unroll
    for (int o = 0; o < C8; ++o) { q[o] = bq[o]; k[o] = bk[o]; }

    for (int ci = 0; ci < C_IN; ++ci) {
        float vf = xf[((b * C_IN + ci) << 12) + pos];
        float vx = x [((b * C_IN + ci) << 12) + pos];
#pragma unroll
        for (int o = 0; o < C8; ++o) {
            q[o] += Wq[o * C_IN + ci] * vf;
            k[o] += Wk[o * C_IN + ci] * vx;
        }
    }

    int t = pos >> 4;
    int m = pos & 15;
    uint32_t* kb = kfrag + (((b << 8) + t) * 32) * 8;
    uint32_t* qb = qfrag + (((b << 8) + t) * 32) * 8;

#pragma unroll
    for (int v = 0; v < 4; ++v) {
        kb[m * 8 + v]            = pack2(k[2 * v],     k[2 * v + 1]);
        kb[m * 8 + 4 + v]        = pack2(k[8 + 2 * v], k[8 + 2 * v + 1]);
        kb[(m + 16) * 8 + v]     = 0u;
        kb[(m + 16) * 8 + 4 + v] = 0u;
    }
#pragma unroll
    for (int v = 0; v < 8; ++v) {
        qb[m * 8 + v]        = pack2(q[2 * v], q[2 * v + 1]);
        qb[(m + 16) * 8 + v] = 0u;
    }
}

// ---------------------------------------------------------------------------
// Kernel 2: value = Wv * x_backward + bv, stored [b][pos][c] (f32).
// x_backward tile (128 rows x 32 cols, row stride 4096) staged into LDS by
// the Tensor Data Mover when available (one wave-level tensor_load_to_lds,
// tracked via TENSORcnt), else by a manual coalesced loop.
// ---------------------------------------------------------------------------
__global__ void proj_v_kernel(const float* __restrict__ xb,
                              const float* __restrict__ Wv,
                              const float* __restrict__ bv,
                              float* __restrict__ val) {
    __shared__ float lds[C_IN * 32];
    int b    = blockIdx.x >> 7;
    int pos0 = (blockIdx.x & 127) * 32;
    int tid  = threadIdx.x;

#if __has_builtin(__builtin_amdgcn_tensor_load_to_lds)
    if (tid < 32) {   // TDM op is wave-level (EXEC ignored); issue from wave 0
        uint64_t gaddr = (uint64_t)(uintptr_t)(xb + (((size_t)b * C_IN) << 12) + pos0);
        uint32_t laddr = (uint32_t)(uintptr_t)&lds[0];
        u32x4 g0 = {};
        g0[0] = 1u;                                        // count=1, user D#
        g0[1] = laddr;                                     // lds_addr
        g0[2] = (uint32_t)gaddr;                           // global_addr[31:0]
        g0[3] = (uint32_t)((gaddr >> 32) & 0x1FFFFFFu) | (2u << 30); // addr[56:32] | type=2
        i32x8 g1 = {};
        g1[0] = (int)(2u << 16);        // workgroup_mask=0, data_size=4B
        g1[1] = (int)(4096u << 16);     // tensor_dim0[15:0] in bits[31:16]
        g1[2] = (int)(128u << 16);      // tensor_dim0 hi=0 | tensor_dim1=128
        g1[3] = (int)(32u << 16);       // tensor_dim1 hi=0 | tile_dim0=32
        g1[4] = 128;                    // tile_dim1=128 (tile_dim2=0)
        g1[5] = 4096;                   // tensor_dim0_stride lo32
        g1[6] = 0; g1[7] = 0;           // stride hi / dim1_stride
        i32x4 z4 = {};
#if __clang_major__ >= 23
        i32x8 z8 = {};
        __builtin_amdgcn_tensor_load_to_lds(g0, g1, z4, z4, z8, 0);
#else
        __builtin_amdgcn_tensor_load_to_lds(g0, g1, z4, z4, 0);
#endif
        __builtin_amdgcn_s_wait_tensorcnt(0);
    }
#else
    for (int e = tid; e < C_IN * 32; e += 256) {
        int ci = e >> 5, p = e & 31;
        lds[ci * 32 + p] = xb[((b * C_IN + ci) << 12) + pos0 + p];
    }
#endif
    __syncthreads();

    int p  = tid & 31;
    int c0 = tid >> 5;
    for (int c = c0; c < C_IN; c += 8) {
        float s = bv[c];
        const float* wr = Wv + c * C_IN;
        for (int ci = 0; ci < C_IN; ++ci)
            s += wr[ci] * lds[ci * 32 + p];
        val[((b << 12) + pos0 + p) * C_IN + c] = s;
    }
}

// ---------------------------------------------------------------------------
// Kernel 3: pack Wf into f16 A-fragments: [cotile(8)][kchunk(72)][lane][8 dw]
// K ordering: K = (dy*3+dx)*256 + ci
// ---------------------------------------------------------------------------
__global__ void prep_wf_kernel(const float* __restrict__ Wf,
                               uint32_t* __restrict__ wfrag) {
    int tid = blockIdx.x * blockDim.x + threadIdx.x;   // 8*72*32 = 18432
    if (tid >= 8 * KCH * 32) return;
    int lane = tid & 31;
    int kc   = (tid >> 5) % KCH;
    int ct   = tid / (KCH * 32);
    int m = lane & 15, khalf = lane >> 4;
    int co = ct * 16 + m;

    uint32_t* dst = wfrag + (((ct * KCH) + kc) * 32 + lane) * 8;
#pragma unroll
    for (int v = 0; v < 8; ++v) {
        float f[2];
#pragma unroll
        for (int pth = 0; pth < 2; ++pth) {
            int e = v * 2 + pth;
            int K = kc * 32 + khalf * 16 + e;
            int tap = K >> 8;          // dy*3+dx
            int ci  = K & 255;
            f[pth] = Wf[(co * 256 + ci) * 9 + tap];
        }
        dst[v] = pack2(f[0], f[1]);
    }
}

// ---------------------------------------------------------------------------
// Kernel 4: attention max/argmax. One wave per k-tile; 2 q-tiles per
// iteration (independent accumulators fill the WMMA->VALU hazard window).
// ---------------------------------------------------------------------------
__global__ void attn_kernel(const uint32_t* __restrict__ kfrag,
                            const uint32_t* __restrict__ qfrag,
                            float* __restrict__ maxv,
                            int* __restrict__ idxv) {
    int lane = threadIdx.x & 31;
    int wid  = threadIdx.x >> 5;
    int ktg  = blockIdx.x * 8 + wid;
    int b    = ktg >> 8;
    int kt   = ktg & 255;

    v16h a = *(const v16h*)(kfrag + (((b << 8) + kt) * 32 + lane) * 8);

    float best[8];
    int   bidx[8];
#pragma unroll
    for (int r = 0; r < 8; ++r) { best[r] = -3.4e38f; bidx[r] = 0; }

    int nq = lane & 15;
    const uint32_t* qp = qfrag + (((b << 8)) * 32 + lane) * 8;
    for (int qt = 0; qt < NTILE; qt += 2) {
        __builtin_prefetch(qp + 1024, 0, 0);          // 4 tiles ahead
        v16h b0 = *(const v16h*)(qp);
        v16h b1 = *(const v16h*)(qp + 256);
        qp += 512;
        v8f c0 = {}, c1 = {};
        c0 = __builtin_amdgcn_wmma_f32_16x16x32_f16(false, a, false, b0,
                                                    (short)0, c0, false, false);
        c1 = __builtin_amdgcn_wmma_f32_16x16x32_f16(false, a, false, b1,
                                                    (short)0, c1, false, false);
        int q0 = qt * 16 + nq;
#pragma unroll
        for (int r = 0; r < 8; ++r) {
            float s = c0[r];
            if (s > best[r]) { best[r] = s; bidx[r] = q0; }
        }
#pragma unroll
        for (int r = 0; r < 8; ++r) {
            float s = c1[r];
            if (s > best[r]) { best[r] = s; bidx[r] = q0 + 16; }
        }
    }

#pragma unroll
    for (int r = 0; r < 8; ++r) {
        float bv = best[r];
        int   bi = bidx[r];
        for (int off = 1; off < 16; off <<= 1) {
            float ov = __shfl_xor(bv, off, 16);
            int   oi = __shfl_xor(bi, off, 16);
            if (ov > bv || (ov == bv && oi < bi)) { bv = ov; bi = oi; }
        }
        if (nq == 0) {
            int row = kt * 16 + r + ((lane >> 4) << 3);
            maxv[(b << 12) + row] = bv;
            idxv[(b << 12) + row] = bi;
        }
    }
}

// ---------------------------------------------------------------------------
// Kernel 5: 3x3 conv (implicit GEMM, f16 WMMA) fused with gather + epilogue.
// Patch in LDS as [row 3][x 18][ci 256 padded to 264] halfs, so each lane's
// 16 contiguous K-elements load as 2 x ds_load_b128.
// ---------------------------------------------------------------------------
__global__ void conv3x3_kernel(const float* __restrict__ x,
                               const float* __restrict__ val,
                               const int* __restrict__ idxv,
                               const float* __restrict__ maxv,
                               const uint32_t* __restrict__ wfrag,
                               const float* __restrict__ bf,
                               float* __restrict__ out) {
    __shared__ _Float16 patch[3 * 18 * CPAD];   // 28512 B

    int bx = blockIdx.x;
    int xt = bx & 3;
    int y  = (bx >> 2) & 63;
    int b  = bx >> 8;
    int x0 = xt * 16;
    int tid = threadIdx.x;

    // --- phase A: x channels (ci 0..127), x-fastest for coalesced reads ---
    for (int f = tid; f < 3 * C_IN * 18; f += 256) {
        int j    = f % 18;
        int rest = f / 18;
        int ci   = rest & 127;
        int rr   = rest >> 7;
        int gy = y - 1 + rr;
        int gx = x0 - 1 + j;
        float v = 0.f;
        if (gy >= 0 && gy < 64 && gx >= 0 && gx < 64)
            v = x[((b * C_IN + ci) << 12) + gy * 64 + gx];
        patch[(rr * 18 + j) * CPAD + ci] = (_Float16)v;
    }
    // --- phase B: gathered value channels (ci 128..255), ci-fastest ---
    for (int f = tid; f < 3 * 18 * C_IN; f += 256) {
        int ci   = f & 127;
        int rest = f >> 7;
        int j    = rest % 18;
        int rr   = rest / 18;
        int gy = y - 1 + rr;
        int gx = x0 - 1 + j;
        float v = 0.f;
        if (gy >= 0 && gy < 64 && gx >= 0 && gx < 64) {
            int gi = idxv[(b << 12) + gy * 64 + gx];
            v = val[((b << 12) + gi) * C_IN + ci];
        }
        patch[(rr * 18 + j) * CPAD + C_IN + ci] = (_Float16)v;
    }
    __syncthreads();

    // --- implicit GEMM: wave w -> co-tile w ---
    int lane  = tid & 31;
    int wid   = tid >> 5;
    int nl    = lane & 15;
    int khalf = lane >> 4;

    const uint32_t* wfp = wfrag + ((wid * KCH) * 32 + lane) * 8;
    v8f acc = {};
    for (int tap = 0; tap < 9; ++tap) {           // (dy,dx)
        int dy = tap / 3;                         // compile-time after unroll? keep simple
        int dx = tap - dy * 3;
        const _Float16* prow = &patch[(dy * 18 + nl + dx) * CPAD + khalf * 16];
#pragma unroll
        for (int k8 = 0; k8 < 8; ++k8) {          // 8 K-chunks per tap
            __builtin_prefetch(wfp + 256, 0, 0);
            v16h a = *(const v16h*)(wfp);
            wfp += 256;
            union { v16h v; uint4 q[2]; } bu;
            const _Float16* src = prow + k8 * 32;
            bu.q[0] = *(const uint4*)(src);
            bu.q[1] = *(const uint4*)(src + 8);
            acc = __builtin_amdgcn_wmma_f32_16x16x32_f16(false, a, false, bu.v,
                                                         (short)0, acc, false, false);
        }
    }

    // --- epilogue: out = x + (acc + bf) * max_value ---
    int pos = y * 64 + x0 + nl;
    float mv = maxv[(b << 12) + pos];
#pragma unroll
    for (int r = 0; r < 8; ++r) {
        int co = wid * 16 + r + (khalf << 3);
        int gidx = ((b * C_IN + co) << 12) + pos;
        out[gidx] = x[gidx] + (acc[r] + bf[co]) * mv;
    }
}

// ---------------------------------------------------------------------------
extern "C" void kernel_launch(void* const* d_in, const int* in_sizes, int n_in,
                              void* d_out, int out_size, void* d_ws, size_t ws_size,
                              hipStream_t stream) {
    const float* x  = (const float*)d_in[0];
    const float* xf = (const float*)d_in[1];
    const float* xb = (const float*)d_in[2];
    const float* Wq = (const float*)d_in[3];
    const float* bq = (const float*)d_in[4];
    const float* Wk = (const float*)d_in[5];
    const float* bk = (const float*)d_in[6];
    const float* Wv = (const float*)d_in[7];
    const float* bv = (const float*)d_in[8];
    const float* Wf = (const float*)d_in[9];
    const float* bf = (const float*)d_in[10];
    float* out = (float*)d_out;

    char* ws = (char*)d_ws;
    uint32_t* kfrag = (uint32_t*)(ws);
    uint32_t* qfrag = (uint32_t*)(ws + (1u << 20));
    float*    val   = (float*)   (ws + (2u << 20));
    float*    maxv  = (float*)   (ws + 10485760u);
    int*      idxv  = (int*)     (ws + 10551296u);
    uint32_t* wfrag = (uint32_t*)(ws + 10616832u);

    proj_qk_kernel<<<64, 256, 0, stream>>>(x, xf, Wq, bq, Wk, bk, kfrag, qfrag);
    proj_v_kernel<<<512, 256, 0, stream>>>(xb, Wv, bv, val);
    prep_wf_kernel<<<72, 256, 0, stream>>>(Wf, wfrag);
    attn_kernel<<<128, 256, 0, stream>>>(kfrag, qfrag, maxv, idxv);
    conv3x3_kernel<<<1024, 256, 0, stream>>>(x, val, idxv, maxv, wfrag, bf, out);
}
